// NeuroSymbolicConfidenceModel_85959475462287
// MI455X (gfx1250) — compile-verified, matrix operand order checked by reading, not run on hardware
//
#include <hip/hip_runtime.h>
#include <math.h>

typedef __attribute__((ext_vector_type(16))) _Float16 v16h;
typedef __attribute__((ext_vector_type(8)))  _Float16 v8h;
typedef __attribute__((ext_vector_type(4)))  _Float16 v4h;
typedef __attribute__((ext_vector_type(8)))  float    v8f;

#define NN   50000
#define EE   800000
#define BB   64
#define HH   128
#define LL   3

#define WMMA_F16(a, b, c) __builtin_amdgcn_wmma_f32_16x16x32_f16(false, (a), false, (b), (short)0, (c), false, false)

// ---------------------------------------------------------------------------
// WMMA tile helpers (v_wmma_f32_16x16x32_f16, wave32), pure-f16 operands.
// A (16x32 f16): lane<16 -> row=lane,    K in {k0+0..7, k0+16..23}
//                lane>=16 -> row=lane-16, K in {k0+8..15, k0+24..31}
// B (32x16 f16): same pattern with N (column) in place of M (B = W^T, W row-major).
// C/D (16x16 f32): elem r -> row = r + 8*(lane>>4), col = lane&15.
// ---------------------------------------------------------------------------
__device__ __forceinline__ v16h load_tile16(const _Float16* __restrict__ P, int r0, int k0, int lane) {
  int r  = r0 + (lane & 15);
  int kb = k0 + ((lane >> 4) << 3);
  const _Float16* p = P + (size_t)r * HH + kb;
  v8h lo = *(const v8h*)p;            // 16B: K kb..kb+7
  v8h hi = *(const v8h*)(p + 16);     // 16B: K kb+16..kb+23
  v16h a;
#pragma unroll
  for (int i = 0; i < 8; ++i) { a[i] = lo[i]; a[8 + i] = hi[i]; }
  return a;
}

__device__ __forceinline__ void store_tile(float* __restrict__ Out, _Float16* __restrict__ OutH,
                                           int row0, int col0, int lane, v8f c,
                                           const float* __restrict__ bias, int do_relu) {
  int cc = col0 + (lane & 15);
  int rb = row0 + ((lane >> 4) << 3);
  float bv = bias ? bias[cc] : 0.0f;
#pragma unroll
  for (int r = 0; r < 8; ++r) {
    float vv = c[r] + bv;
    if (do_relu) vv = fmaxf(vv, 0.0f);
    Out[(size_t)(rb + r) * HH + cc] = vv;
    if (OutH) OutH[(size_t)(rb + r) * HH + cc] = (_Float16)vv;
  }
}

// ---------------------------------------------------------------------------
// GEMM: Out[M,128] = act(A[M,128] @ W[128,128]^T + bias), all-f16 operand streams.
// block = 256 thr (8 waves) = 32 rows x 128 cols; each wave: 16 rows x 32 cols (2 acc).
// ---------------------------------------------------------------------------
__global__ void wmma_gemm_f16_kernel(const _Float16* __restrict__ A, const _Float16* __restrict__ Wh,
                                     const float* __restrict__ bias, float* __restrict__ Out,
                                     _Float16* __restrict__ OutH, int M, int do_relu) {
  int wave = threadIdx.x >> 5, lane = threadIdx.x & 31;
  int row = blockIdx.x * 32 + (wave >> 2) * 16;
  int col = (wave & 3) * 32;
  if (row >= M) return;
  v8f acc0 = {}, acc1 = {};
#pragma unroll
  for (int k = 0; k < HH; k += 32) {
    v16h a  = load_tile16(A, row, k, lane);
    v16h b0 = load_tile16(Wh, col, k, lane);
    v16h b1 = load_tile16(Wh, col + 16, k, lane);
    acc0 = WMMA_F16(a, b0, acc0);
    acc1 = WMMA_F16(a, b1, acc1);
  }
  store_tile(Out, OutH, row, col, lane, acc0, bias, do_relu);
  store_tile(Out, OutH, row, col + 16, lane, acc1, bias, do_relu);
}

// h = aggh @ Wl^T + bl + xh @ Wr^T  (aggh already inv_deg-scaled f16)
__global__ void sage_wmma_f16_kernel(const _Float16* __restrict__ aggh, const _Float16* __restrict__ xh,
                                     const _Float16* __restrict__ Wlh, const float* __restrict__ bl,
                                     const _Float16* __restrict__ Wrh, float* __restrict__ hOut, int M) {
  int wave = threadIdx.x >> 5, lane = threadIdx.x & 31;
  int row = blockIdx.x * 32 + (wave >> 2) * 16;
  int col = (wave & 3) * 32;
  if (row >= M) return;
  v8f acc0 = {}, acc1 = {};
#pragma unroll
  for (int k = 0; k < HH; k += 32) {
    v16h a1 = load_tile16(aggh, row, k, lane);
    v16h b0 = load_tile16(Wlh, col, k, lane);
    v16h b1 = load_tile16(Wlh, col + 16, k, lane);
    acc0 = WMMA_F16(a1, b0, acc0);
    acc1 = WMMA_F16(a1, b1, acc1);
    v16h a2 = load_tile16(xh, row, k, lane);
    v16h c0 = load_tile16(Wrh, col, k, lane);
    v16h c1 = load_tile16(Wrh, col + 16, k, lane);
    acc0 = WMMA_F16(a2, c0, acc0);
    acc1 = WMMA_F16(a2, c1, acc1);
  }
  store_tile(hOut, nullptr, row, col, lane, acc0, bl, 0);
  store_tile(hOut, nullptr, row, col + 16, lane, acc1, bl, 0);
}

// ---------------------------------------------------------------------------
// Conversion / fill kernels
// ---------------------------------------------------------------------------
__global__ void fill4_kernel(float* __restrict__ p, float v, size_t n4) {
  size_t i = (size_t)blockIdx.x * blockDim.x + threadIdx.x;
  if (i < n4) { float4 o = {v, v, v, v}; ((float4*)p)[i] = o; }
}

// 8 elements per thread
__global__ void cvt_f16_kernel(const float* __restrict__ src, _Float16* __restrict__ dst, size_t n8) {
  size_t i = (size_t)blockIdx.x * blockDim.x + threadIdx.x;
  if (i >= n8) return;
  const float4* s = (const float4*)src + i * 2;
  float4 a = s[0], b = s[1];
  v8h o;
  o[0] = (_Float16)a.x; o[1] = (_Float16)a.y; o[2] = (_Float16)a.z; o[3] = (_Float16)a.w;
  o[4] = (_Float16)b.x; o[5] = (_Float16)b.y; o[6] = (_Float16)b.z; o[7] = (_Float16)b.w;
  ((v8h*)dst)[i] = o;
}

// aggh[n,:] = (f16)(agg[n,:] * inv[n]); 8 elems/thread, 16 threads per row
__global__ void scale_cvt_kernel(const float* __restrict__ agg, const float* __restrict__ inv,
                                 _Float16* __restrict__ aggh, int Nn) {
  int i = blockIdx.x * blockDim.x + threadIdx.x;
  if (i >= Nn * 16) return;
  float sc = inv[i >> 4];
  const float4* s = (const float4*)agg + (size_t)i * 2;
  float4 a = s[0], b = s[1];
  v8h o;
  o[0] = (_Float16)(a.x * sc); o[1] = (_Float16)(a.y * sc); o[2] = (_Float16)(a.z * sc); o[3] = (_Float16)(a.w * sc);
  o[4] = (_Float16)(b.x * sc); o[5] = (_Float16)(b.y * sc); o[6] = (_Float16)(b.z * sc); o[7] = (_Float16)(b.w * sc);
  ((v8h*)aggh)[i] = o;
}

// ---------------------------------------------------------------------------
// Graph / segment kernels
// ---------------------------------------------------------------------------
__global__ void deg_kernel(const int* __restrict__ dst, float* __restrict__ deg, int E) {
  int e = blockIdx.x * blockDim.x + threadIdx.x;
  if (e < E) atomicAdd(&deg[dst[e]], 1.0f);
}

__global__ void invdeg_kernel(const float* __restrict__ deg, float* __restrict__ inv, int n) {
  int i = blockIdx.x * blockDim.x + threadIdx.x;
  if (i < n) inv[i] = 1.0f / fmaxf(deg[i], 1.0f);
}

// 32 lanes per edge, 4 floats per lane: agg[dst] += x[src]
__global__ void edge_scatter_kernel(const int* __restrict__ src, const int* __restrict__ dst,
                                    const float* __restrict__ x, float* __restrict__ agg, int E) {
  unsigned tid = blockIdx.x * blockDim.x + threadIdx.x;
  int e = tid >> 5, lane = tid & 31;
  if (e >= E) return;
  int s = src[e], d = dst[e];
  float4 v = ((const float4*)(x + (size_t)s * HH))[lane];
  float* o = agg + (size_t)d * HH + lane * 4;
  atomicAdd(o + 0, v.x); atomicAdd(o + 1, v.y);
  atomicAdd(o + 2, v.z); atomicAdd(o + 3, v.w);
}

// x = relu(LN(h)*g + beta) + x ; also writes f16 mirror xh. One wave per row.
__global__ void ln_relu_res_kernel(const float* __restrict__ h, const float* __restrict__ g,
                                   const float* __restrict__ beta, float* __restrict__ x,
                                   _Float16* __restrict__ xh, int M) {
  int row = blockIdx.x * (blockDim.x >> 5) + (threadIdx.x >> 5);
  int lane = threadIdx.x & 31;
  if (row >= M) return;
  float4 hv = ((const float4*)(h + (size_t)row * HH))[lane];
  float s = hv.x + hv.y + hv.z + hv.w;
#pragma unroll
  for (int off = 16; off; off >>= 1) s += __shfl_xor(s, off, 32);
  float mean = s * (1.0f / 128.0f);
  float dx = hv.x - mean, dy = hv.y - mean, dz = hv.z - mean, dw = hv.w - mean;
  float ss = dx * dx + dy * dy + dz * dz + dw * dw;
#pragma unroll
  for (int off = 16; off; off >>= 1) ss += __shfl_xor(ss, off, 32);
  float rstd = rsqrtf(ss * (1.0f / 128.0f) + 1e-5f);
  int c = lane * 4;
  float4 res = ((const float4*)(x + (size_t)row * HH))[lane];
  float4 o;
  o.x = fmaxf(dx * rstd * g[c + 0] + beta[c + 0], 0.0f) + res.x;
  o.y = fmaxf(dy * rstd * g[c + 1] + beta[c + 1], 0.0f) + res.y;
  o.z = fmaxf(dz * rstd * g[c + 2] + beta[c + 2], 0.0f) + res.z;
  o.w = fmaxf(dw * rstd * g[c + 3] + beta[c + 3], 0.0f) + res.w;
  ((float4*)(x + (size_t)row * HH))[lane] = o;
  v4h oh = {(_Float16)o.x, (_Float16)o.y, (_Float16)o.z, (_Float16)o.w};
  *(v4h*)(xh + (size_t)row * HH + c) = oh;
}

// Out = LN(In)*g + beta (plain, one wave per row)
__global__ void ln_plain_kernel(const float* __restrict__ In, const float* __restrict__ g,
                                const float* __restrict__ beta, float* __restrict__ Out, int M) {
  int row = blockIdx.x * (blockDim.x >> 5) + (threadIdx.x >> 5);
  int lane = threadIdx.x & 31;
  if (row >= M) return;
  float4 hv = ((const float4*)(In + (size_t)row * HH))[lane];
  float s = hv.x + hv.y + hv.z + hv.w;
#pragma unroll
  for (int off = 16; off; off >>= 1) s += __shfl_xor(s, off, 32);
  float mean = s * (1.0f / 128.0f);
  float dx = hv.x - mean, dy = hv.y - mean, dz = hv.z - mean, dw = hv.w - mean;
  float ss = dx * dx + dy * dy + dz * dz + dw * dw;
#pragma unroll
  for (int off = 16; off; off >>= 1) ss += __shfl_xor(ss, off, 32);
  float rstd = rsqrtf(ss * (1.0f / 128.0f) + 1e-5f);
  int c = lane * 4;
  float4 o;
  o.x = dx * rstd * g[c + 0] + beta[c + 0];
  o.y = dy * rstd * g[c + 1] + beta[c + 1];
  o.z = dz * rstd * g[c + 2] + beta[c + 2];
  o.w = dw * rstd * g[c + 3] + beta[c + 3];
  ((float4*)(Out + (size_t)row * HH))[lane] = o;
}

// ---------------------------------------------------------------------------
// Attention pooling
// ---------------------------------------------------------------------------
__global__ void qvec_kernel(const float* __restrict__ inW, const float* __restrict__ inB,
                            const float* __restrict__ query, float* __restrict__ qvec) {
  int h = threadIdx.x;
  const float* wr = inW + (size_t)h * HH;
  float s = inB[h];
  for (int j = 0; j < HH; ++j) s += wr[j] * query[j];
  qvec[h] = s;
}

// fold q into Wk: s_w[head][j] = sum_d Wk[head*32+d, j]*q[head*32+d]; s_b[head] = bk.q
__global__ void sw_kernel(const float* __restrict__ inW, const float* __restrict__ inB,
                          const float* __restrict__ qvec, float* __restrict__ s_w, float* __restrict__ s_b) {
  int tid = blockIdx.x * blockDim.x + threadIdx.x;
  const float* Wk = inW + 128 * 128;
  const float* bk = inB + 128;
  if (tid < 512) {
    int hd = tid >> 7, j = tid & 127;
    float s = 0.0f;
    for (int d = 0; d < 32; ++d) s += Wk[(size_t)(hd * 32 + d) * HH + j] * qvec[hd * 32 + d];
    s_w[hd * HH + j] = s;
  } else if (tid < 516) {
    int hd = tid - 512;
    float s = 0.0f;
    for (int d = 0; d < 32; ++d) s += bk[hd * 32 + d] * qvec[hd * 32 + d];
    s_b[hd] = s;
  }
}

__device__ __forceinline__ void atomicMaxFloat(float* addr, float val) {
  if (val >= 0.0f) atomicMax((int*)addr, __float_as_int(val));
  else             atomicMin((unsigned int*)addr, (unsigned int)__float_as_int(val));
}

__global__ void scores_kernel(const float* __restrict__ x, const float* __restrict__ s_w,
                              const float* __restrict__ s_b, const int* __restrict__ batch,
                              float* __restrict__ scores, float* __restrict__ smax, int Nn) {
  int tid = blockIdx.x * blockDim.x + threadIdx.x;
  if (tid >= Nn * 4) return;
  int n = tid >> 2, hd = tid & 3;
  const float* xr = x + (size_t)n * HH;
  const float* wr = s_w + hd * HH;
  float s = 0.0f;
  for (int j = 0; j < HH; ++j) s += xr[j] * wr[j];
  s = (s + s_b[hd]) * 0.17677669529663687f;   // 1/sqrt(32)
  scores[tid] = s;
  atomicMaxFloat(&smax[batch[n] * 4 + hd], s);
}

__global__ void expden_kernel(const float* __restrict__ scores, const float* __restrict__ smax,
                              const int* __restrict__ batch, float* __restrict__ eexp,
                              float* __restrict__ den, int Nn) {
  int tid = blockIdx.x * blockDim.x + threadIdx.x;
  if (tid >= Nn * 4) return;
  int n = tid >> 2, hd = tid & 3;
  int b = batch[n];
  float e = expf(scores[tid] - smax[b * 4 + hd]);
  eexp[tid] = e;
  atomicAdd(&den[b * 4 + hd], e);
}

__global__ void pooled_kernel(const float* __restrict__ v, const float* __restrict__ eexp,
                              const float* __restrict__ den, const int* __restrict__ batch,
                              float* __restrict__ pooled, int Nn) {
  int tid = blockIdx.x * blockDim.x + threadIdx.x;
  if (tid >= Nn * HH) return;
  int n = tid >> 7, j = tid & 127;
  int b = batch[n];
  int hd = j >> 5;
  float w = eexp[n * 4 + hd] / den[b * 4 + hd];
  atomicAdd(&pooled[b * HH + j], w * v[(size_t)n * HH + j]);
}

// ---------------------------------------------------------------------------
// Symbolic processor, fusion, heads (small: B=64)
// ---------------------------------------------------------------------------
__global__ void sym1_kernel(const float* __restrict__ symf, const float* __restrict__ sym_W,
                            const float* __restrict__ sym_b, float* __restrict__ embf) {
  int tid = blockIdx.x * blockDim.x + threadIdx.x;      // B*4*32 = 8192
  if (tid >= BB * 128) return;
  int b = tid >> 7, f = (tid >> 5) & 3, o = tid & 31;
  float s = sym_b[f * 32 + o];
  const float* p = symf + b * 64 + f * 16;
  const float* w = sym_W + (size_t)(f * 32 + o) * 16;
  for (int i = 0; i < 16; ++i) s += p[i] * w[i];
  embf[b * HH + f * 32 + o] = fmaxf(s, 0.0f);
}

__global__ void dense128_relu_kernel(const float* __restrict__ In, const float* __restrict__ W,
                                     const float* __restrict__ bias, float* __restrict__ Out) {
  int tid = blockIdx.x * blockDim.x + threadIdx.x;      // B*128
  if (tid >= BB * HH) return;
  int b = tid >> 7, h = tid & 127;
  const float* w = W + (size_t)h * HH;
  const float* in = In + (size_t)b * HH;
  float s = bias[h];
  for (int j = 0; j < HH; ++j) s += in[j] * w[j];
  Out[tid] = fmaxf(s, 0.0f);
}

__global__ void fusion_kernel(const float* __restrict__ g, const float* __restrict__ sym,
                              const float* __restrict__ fus_W, const float* __restrict__ fus_b,
                              float* __restrict__ Out) {
  int tid = blockIdx.x * blockDim.x + threadIdx.x;      // B*128
  if (tid >= BB * HH) return;
  int b = tid >> 7, h = tid & 127;
  const float* w = fus_W + (size_t)h * 256;
  float s = fus_b[h];
  for (int j = 0; j < HH; ++j) s += g[b * HH + j] * w[j];
  for (int j = 0; j < HH; ++j) s += sym[b * HH + j] * w[128 + j];
  Out[tid] = fmaxf(s, 0.0f);
}

__global__ void head1_kernel(const float* __restrict__ fused, const float* __restrict__ W1,
                             const float* __restrict__ b1, float* __restrict__ hh) {
  int tid = blockIdx.x * blockDim.x + threadIdx.x;      // 3*64*64
  if (tid >= 3 * BB * 64) return;
  int k = tid / (BB * 64), rem = tid % (BB * 64);
  int b = rem >> 6, o = rem & 63;
  const float* w = W1 + ((size_t)k * 64 + o) * HH;
  const float* f = fused + (size_t)b * HH;
  float s = b1[k * 64 + o];
  for (int i = 0; i < HH; ++i) s += f[i] * w[i];
  hh[tid] = fmaxf(s, 0.0f);
}

__global__ void head2_kernel(const float* __restrict__ hh, const float* __restrict__ W2,
                             const float* __restrict__ b2, float* __restrict__ out) {
  int tid = blockIdx.x * blockDim.x + threadIdx.x;      // 3*64
  if (tid >= 3 * BB) return;
  int k = tid >> 6, b = tid & 63;
  const float* w = W2 + k * 64;
  const float* h = hh + (size_t)k * BB * 64 + b * 64;
  float s = b2[k];
  for (int o = 0; o < 64; ++o) s += h[o] * w[o];
  out[k * BB + b] = 1.0f / (1.0f + expf(-s));
}

// ---------------------------------------------------------------------------
extern "C" void kernel_launch(void* const* d_in, const int* in_sizes, int n_in,
                              void* d_out, int out_size, void* d_ws, size_t ws_size,
                              hipStream_t stream) {
  const float* node_features = (const float*)d_in[0];
  const float* sym_features  = (const float*)d_in[1];
  const int*   edge_index    = (const int*)d_in[2];
  const int*   batch         = (const int*)d_in[3];
  const float* W_in   = (const float*)d_in[4];
  const float* b_in   = (const float*)d_in[5];
  const float* sage_Wl = (const float*)d_in[6];
  const float* sage_bl = (const float*)d_in[7];
  const float* sage_Wr = (const float*)d_in[8];
  const float* ln_g = (const float*)d_in[9];
  const float* ln_b = (const float*)d_in[10];
  const float* query = (const float*)d_in[11];
  const float* in_proj_W = (const float*)d_in[12];
  const float* in_proj_b = (const float*)d_in[13];
  const float* out_W = (const float*)d_in[14];
  const float* out_b = (const float*)d_in[15];
  const float* sym_W = (const float*)d_in[16];
  const float* sym_b = (const float*)d_in[17];
  const float* symf_W = (const float*)d_in[18];
  const float* symf_b = (const float*)d_in[19];
  const float* symf_g = (const float*)d_in[20];
  const float* symf_beta = (const float*)d_in[21];
  const float* fus_W = (const float*)d_in[22];
  const float* fus_b = (const float*)d_in[23];
  const float* fus_g = (const float*)d_in[24];
  const float* fus_beta = (const float*)d_in[25];
  const float* head_W1 = (const float*)d_in[26];
  const float* head_b1 = (const float*)d_in[27];
  const float* head_W2 = (const float*)d_in[28];
  const float* head_b2 = (const float*)d_in[29];
  float* out = (float*)d_out;

  const int* src = edge_index;
  const int* dst = edge_index + EE;

  // ---- workspace carve-up: f32 region, then f16 region ----
  float* ws = (float*)d_ws;
  size_t off = 0;
  auto alloc = [&](size_t n) { float* p = ws + off; off += n; return p; };
  float* x     = alloc((size_t)NN * HH);
  float* agg   = alloc((size_t)NN * HH);
  float* hbuf  = alloc((size_t)NN * HH);   // reused as v after SAGE layers
  float* deg   = alloc(NN);
  float* inv   = alloc(NN);
  float* scores = alloc((size_t)NN * 4);
  float* eexp   = alloc((size_t)NN * 4);
  float* qvec = alloc(HH);
  float* s_w  = alloc(4 * HH);
  float* s_b  = alloc(4);
  float* smax = alloc(BB * 4);
  float* den  = alloc(BB * 4);
  float* pooled = alloc(BB * HH);
  float* gbuf   = alloc(BB * HH);
  float* embf   = alloc(BB * HH);
  float* t1     = alloc(BB * HH);
  float* symo   = alloc(BB * HH);
  float* fpre   = alloc(BB * HH);
  float* fused  = alloc(BB * HH);
  float* hh     = alloc((size_t)3 * BB * 64);
  off = (off + 7) & ~(size_t)7;            // 16B-align the f16 region
  _Float16* hbase = (_Float16*)(ws + off);
  size_t hoff = 0;
  auto halloc = [&](size_t n) { _Float16* p = hbase + hoff; hoff += (n + 7) & ~(size_t)7; return p; };
  _Float16* xh   = halloc((size_t)NN * HH);
  _Float16* ah   = halloc((size_t)NN * HH);   // nfh (pre-layers) then aggh (per layer)
  _Float16* Wh_in  = halloc(HH * HH);
  _Float16* Wh_l   = halloc((size_t)LL * HH * HH);
  _Float16* Wh_r   = halloc((size_t)LL * HH * HH);
  _Float16* Wh_v   = halloc(HH * HH);
  _Float16* Wh_out = halloc(HH * HH);
  _Float16* pooledh = halloc(BB * HH);
  (void)ws_size; (void)n_in; (void)in_sizes; (void)out_size;

  auto fill = [&](float* p, float v, size_t n) {   // n must be multiple of 4
    fill4_kernel<<<(unsigned)((n / 4 + 255) / 256), 256, 0, stream>>>(p, v, n / 4);
  };
  auto cvt = [&](const float* s, _Float16* d, size_t n) {  // n multiple of 8
    cvt_f16_kernel<<<(unsigned)((n / 8 + 255) / 256), 256, 0, stream>>>(s, d, n / 8);
  };

  const int GEMM_GRID_N = (NN + 31) / 32;   // 1563
  const int GEMM_GRID_B = (BB + 31) / 32;   // 2

  // ---- weight pre-conversion to f16 (L2-resident, done once per call) ----
  cvt(W_in, Wh_in, (size_t)HH * HH);
  cvt(sage_Wl, Wh_l, (size_t)LL * HH * HH);
  cvt(sage_Wr, Wh_r, (size_t)LL * HH * HH);
  cvt(in_proj_W + 2 * HH * HH, Wh_v, (size_t)HH * HH);
  cvt(out_W, Wh_out, (size_t)HH * HH);

  // ---- degree + inverse degree ----
  fill(deg, 0.0f, NN);
  deg_kernel<<<(EE + 255) / 256, 256, 0, stream>>>(dst, deg, EE);
  invdeg_kernel<<<(NN + 255) / 256, 256, 0, stream>>>(deg, inv, NN);

  // ---- input projection: x = relu(NF @ W_in^T + b_in) ----
  cvt(node_features, ah, (size_t)NN * HH);                       // ah = nfh
  wmma_gemm_f16_kernel<<<GEMM_GRID_N, 256, 0, stream>>>(ah, Wh_in, b_in, x, xh, NN, 1);

  // ---- 3 SAGE layers ----
  for (int l = 0; l < LL; ++l) {
    fill(agg, 0.0f, (size_t)NN * HH);
    edge_scatter_kernel<<<(unsigned)(((size_t)EE * 32) / 256), 256, 0, stream>>>(src, dst, x, agg, EE);
    scale_cvt_kernel<<<(NN * 16 + 255) / 256, 256, 0, stream>>>(agg, inv, ah, NN);
    sage_wmma_f16_kernel<<<GEMM_GRID_N, 256, 0, stream>>>(ah, xh,
        Wh_l + (size_t)l * HH * HH, sage_bl + (size_t)l * HH,
        Wh_r + (size_t)l * HH * HH, hbuf, NN);
    ln_relu_res_kernel<<<NN / 8, 256, 0, stream>>>(hbuf, ln_g + (size_t)l * HH, ln_b + (size_t)l * HH, x, xh, NN);
  }

  // ---- attention pooling ----
  qvec_kernel<<<1, 128, 0, stream>>>(in_proj_W, in_proj_b, query, qvec);
  sw_kernel<<<3, 256, 0, stream>>>(in_proj_W, in_proj_b, qvec, s_w, s_b);
  fill(smax, -INFINITY, BB * 4);
  fill(den, 0.0f, BB * 4);
  scores_kernel<<<(NN * 4 + 255) / 256, 256, 0, stream>>>(x, s_w, s_b, batch, scores, smax, NN);
  expden_kernel<<<(NN * 4 + 255) / 256, 256, 0, stream>>>(scores, smax, batch, eexp, den, NN);
  float* vbuf = hbuf;   // v = x @ Wv^T + bv
  wmma_gemm_f16_kernel<<<GEMM_GRID_N, 256, 0, stream>>>(xh, Wh_v, in_proj_b + 256, vbuf, nullptr, NN, 0);
  fill(pooled, 0.0f, (size_t)BB * HH);
  pooled_kernel<<<(unsigned)(((size_t)NN * HH + 255) / 256), 256, 0, stream>>>(vbuf, eexp, den, batch, pooled, NN);
  cvt(pooled, pooledh, (size_t)BB * HH);
  wmma_gemm_f16_kernel<<<GEMM_GRID_B, 256, 0, stream>>>(pooledh, Wh_out, out_b, gbuf, nullptr, BB, 0);

  // ---- symbolic processor ----
  sym1_kernel<<<(BB * HH + 255) / 256, 256, 0, stream>>>(sym_features, sym_W, sym_b, embf);
  dense128_relu_kernel<<<(BB * HH + 255) / 256, 256, 0, stream>>>(embf, symf_W, symf_b, t1);
  ln_plain_kernel<<<BB / 8, 256, 0, stream>>>(t1, symf_g, symf_beta, symo, BB);

  // ---- fusion ----
  fusion_kernel<<<(BB * HH + 255) / 256, 256, 0, stream>>>(gbuf, symo, fus_W, fus_b, fpre);
  ln_plain_kernel<<<BB / 8, 256, 0, stream>>>(fpre, fus_g, fus_beta, fused, BB);

  // ---- heads ----
  head1_kernel<<<(3 * BB * 64 + 255) / 256, 256, 0, stream>>>(fused, head_W1, head_b1, hh);
  head2_kernel<<<1, 256, 0, stream>>>(hh, head_W2, head_b2, out);
}
